// RoutingLinear_9869834846658
// MI455X (gfx1250) — matrix operand order repaired and checked by reading, not compiled, not run on hardware
//
#include <hip/hip_runtime.h>

// ---- CDNA5 WMMA types -------------------------------------------------------
typedef __attribute__((ext_vector_type(16))) __bf16 v16bf;
typedef __attribute__((ext_vector_type(8)))  __bf16 bf16x8;
typedef __attribute__((ext_vector_type(4)))  __bf16 bf16x4;
typedef __attribute__((ext_vector_type(8)))  float  v8f;
typedef __attribute__((ext_vector_type(4)))  float  f32x4;

// ---- problem size (fixed by reference) --------------------------------------
#define TOKENS 2048
#define DIM    4096
#define VOCAB  32000

// ---- tiling -----------------------------------------------------------------
#define BM   128          // tokens per block
#define BN   128          // vocab per block
#define BK   32           // k-chunk (one WMMA K)
#define LB   40           // bf16 LDS row stride: 80B = 16B-aligned, 20-bank step
#define NKC  (DIM / BK)   // 128 k-chunks

// fp32 -> bf16 hi/lo split of a float4 (fp32 ~= hi + lo, ~16 mantissa bits)
__device__ __forceinline__ void split4(const f32x4 q, bf16x4& h, bf16x4& l)
{
#pragma unroll
    for (int i = 0; i < 4; ++i) {
        const __bf16 hh = (__bf16)q[i];
        h[i] = hh;
        l[i] = (__bf16)(q[i] - (float)hh);
    }
}

__global__ __launch_bounds__(256)
void routing_linear_bf16x3(const float* __restrict__ x,
                           const float* __restrict__ W,
                           const float* __restrict__ bias,
                           float* __restrict__ out)
{
    // double-buffered bf16 hi/lo tiles (80 KB total; gfx1250 WG limit is 320 KB)
    __shared__ __bf16 sXh[2][BM][LB];   // 20 KB
    __shared__ __bf16 sXl[2][BM][LB];   // 20 KB
    __shared__ __bf16 sWh[2][BN][LB];   // 20 KB
    __shared__ __bf16 sWl[2][BN][LB];   // 20 KB

    const int tid   = threadIdx.x;
    const int lane  = tid & 31;
    const int wave  = tid >> 5;
    const int waveM = wave & 3;    // 4 wave-rows x 32 tokens
    const int waveN = wave >> 2;   // 2 wave-cols x 64 vocab
    const int l16   = lane & 15;
    const int lh    = lane >> 4;   // lane half (0/1)

    const int blockM = blockIdx.x * BM;
    const int blockN = blockIdx.y * BN;

    const float* xg = x + (size_t)blockM * DIM;
    const float* Wg = W + (size_t)blockN * DIM;

    // per-thread tile offsets are loop-invariant 32-bit values -> GVS addressing
    // (scalar base advances with kc; vector offset computed once)
    unsigned off[4];
    int      rowq[4], kqq[4];
#pragma unroll
    for (int i = 0; i < 4; ++i) {
        const int idx = tid + i * 256;            // 1024 f32x4 per 128x32 tile
        rowq[i] = idx >> 3;
        kqq[i]  = (idx & 7) << 2;
        off[i]  = (unsigned)(rowq[i] * DIM + kqq[i]);
    }

    f32x4 rx[4], rw[4];   // register staging (global -> regs -> convert -> LDS)

    auto load_tiles = [&](int kc) {
        const float* xp = xg + kc * BK;           // uniform scalar base
        const float* Wp = Wg + kc * BK;
#pragma unroll
        for (int i = 0; i < 4; ++i) rx[i] = *(const f32x4*)(xp + off[i]);
#pragma unroll
        for (int i = 0; i < 4; ++i) rw[i] = *(const f32x4*)(Wp + off[i]);
    };

    auto convert_store = [&](int buf) {
#pragma unroll
        for (int i = 0; i < 4; ++i) {
            bf16x4 h, l;
            split4(rx[i], h, l);
            *(bf16x4*)&sXh[buf][rowq[i]][kqq[i]] = h;   // 8B ds_store
            *(bf16x4*)&sXl[buf][rowq[i]][kqq[i]] = l;
        }
#pragma unroll
        for (int i = 0; i < 4; ++i) {
            bf16x4 h, l;
            split4(rw[i], h, l);
            *(bf16x4*)&sWh[buf][rowq[i]][kqq[i]] = h;
            *(bf16x4*)&sWl[buf][rowq[i]][kqq[i]] = l;
        }
    };

    // A 16x32 frag: lane l16 = row M; per-lane K = {8lh..+7} u {16+8lh..+7}
    auto frag_a = [&](const __bf16* tile, int mt) -> v16bf {
        const __bf16* p = tile + (waveM * 32 + mt * 16 + l16) * LB + lh * 8;
        const bf16x8 c0 = *(const bf16x8*)(p);        // 16B, aligned
        const bf16x8 c1 = *(const bf16x8*)(p + 16);   // +32B
        return __builtin_shufflevector(c0, c1, 0,1,2,3,4,5,6,7,8,9,10,11,12,13,14,15);
    };
    // B 32x16 frag: lane l16 = col N (= W row); per-lane K = {8lh..+7} u {16+8lh..+7}
    auto frag_b = [&](const __bf16* tile, int nt) -> v16bf {
        const __bf16* p = tile + (waveN * 64 + nt * 16 + l16) * LB + lh * 8;
        const bf16x8 c0 = *(const bf16x8*)(p);
        const bf16x8 c1 = *(const bf16x8*)(p + 16);
        return __builtin_shufflevector(c0, c1, 0,1,2,3,4,5,6,7,8,9,10,11,12,13,14,15);
    };

    v8f acc[2][4] = {};   // fp32 accumulators: 2x4 WMMA tiles per wave (32x64)

    load_tiles(0);
    convert_store(0);
    __syncthreads();

    for (int kc = 0; kc < NKC; ++kc) {
        const int buf = kc & 1;
        if (kc + 1 < NKC) load_tiles(kc + 1);   // HBM loads in flight during WMMA

        // ---- pure ds_load fragment build (WMMA register layout) ----
        v16bf aH[2], aL[2], bH[4], bL[4];
#pragma unroll
        for (int mt = 0; mt < 2; ++mt) {
            aH[mt] = frag_a(&sXh[buf][0][0], mt);
            aL[mt] = frag_a(&sXl[buf][0][0], mt);
        }
#pragma unroll
        for (int nt = 0; nt < 4; ++nt) {
            bH[nt] = frag_b(&sWh[buf][0][0], nt);
            bL[nt] = frag_b(&sWl[buf][0][0], nt);
        }

        // ---- bf16x3 fp32-emulation, grouped so consecutive WMMAs hit
        //      different accumulators (8 independent chains) ----
#pragma unroll
        for (int mt = 0; mt < 2; ++mt)
#pragma unroll
            for (int nt = 0; nt < 4; ++nt)
                acc[mt][nt] = __builtin_amdgcn_wmma_f32_16x16x32_bf16(
                    false, aH[mt], false, bH[nt], (short)0, acc[mt][nt], false, false);
#pragma unroll
        for (int mt = 0; mt < 2; ++mt)
#pragma unroll
            for (int nt = 0; nt < 4; ++nt)
                acc[mt][nt] = __builtin_amdgcn_wmma_f32_16x16x32_bf16(
                    false, aH[mt], false, bL[nt], (short)0, acc[mt][nt], false, false);
#pragma unroll
        for (int mt = 0; mt < 2; ++mt)
#pragma unroll
            for (int nt = 0; nt < 4; ++nt)
                acc[mt][nt] = __builtin_amdgcn_wmma_f32_16x16x32_bf16(
                    false, aL[mt], false, bH[nt], (short)0, acc[mt][nt], false, false);

        __syncthreads();                           // reads of buf complete
        if (kc + 1 < NKC) convert_store(buf ^ 1);  // convert+publish next chunk
        __syncthreads();
    }

    // ---- epilogue: C layout (VGPR g -> M = 8*lh + g, N = l16), add bias ----
#pragma unroll
    for (int nt = 0; nt < 4; ++nt) {
        const int   col = blockN + waveN * 64 + nt * 16 + l16;
        const float bv  = bias[col];
#pragma unroll
        for (int mt = 0; mt < 2; ++mt) {
            const int row0 = blockM + waveM * 32 + mt * 16 + lh * 8;
#pragma unroll
            for (int g = 0; g < 8; ++g) {
                out[(size_t)(row0 + g) * VOCAB + col] = acc[mt][nt][g] + bv;
            }
        }
    }
}

extern "C" void kernel_launch(void* const* d_in, const int* in_sizes, int n_in,
                              void* d_out, int out_size, void* d_ws, size_t ws_size,
                              hipStream_t stream)
{
    (void)in_sizes; (void)n_in; (void)out_size; (void)d_ws; (void)ws_size;
    const float* x = (const float*)d_in[0];   // [2048, 4096]
    const float* W = (const float*)d_in[1];   // [32000, 4096]
    const float* b = (const float*)d_in[2];   // [32000]
    float* out = (float*)d_out;               // [2048, 32000]

    dim3 grid(TOKENS / BM, VOCAB / BN);       // (16, 250): M fastest -> W bands hot in L2
    dim3 block(256);                           // 8 wave32 waves
    routing_linear_bf16x3<<<grid, block, 0, stream>>>(x, W, b, out);
}
